// VQEmbedding_24146306138336
// MI455X (gfx1250) — matrix operand order repaired and checked by reading, not compiled, not run on hardware
//
#include <hip/hip_runtime.h>
#include <hip/hip_bf16.h>

#define B_ 16
#define D_ 256
#define T_ 2048
#define K_ 8192
#define N_ (B_ * T_)          // 32768 rows
#define QOUT (B_ * D_ * T_)   // 8388608 quantized elements
#define ROWS_PER_BLOCK 128
#define NBLOCKS (N_ / ROWS_PER_BLOCK)  // 256
#define GROUP_CODES 64
#define NGROUPS (K_ / GROUP_CODES)     // 128
#define BUF_HALF 32768                 // 64 rows * 512B (one precision half)
#define BUF_BYTES (2 * BUF_HALF)       // hi + lo = 64KB per buffer

// Dynamic LDS: [bufB0 64KB][bufB1 64KB][s_idx 512B][s_red 1KB]
#define LDS_TOTAL (2 * BUF_BYTES + ROWS_PER_BLOCK * 4 + 256 * 4)

typedef __attribute__((ext_vector_type(16))) __bf16 v16bf;
typedef __attribute__((ext_vector_type(8)))  float  v8f;

// Async global->LDS copy of 16 bytes per lane (GLOBAL_LOAD_ASYNC_TO_LDS_B128,
// tracked by ASYNCcnt). LDS offset = low 32 bits of the flat shared address.
__device__ __forceinline__ void async_copy_b128(void* ldsDst, const void* src) {
  unsigned ldsOff = (unsigned)(unsigned long long)(uintptr_t)ldsDst;
  unsigned long long g = (unsigned long long)(uintptr_t)src;
  asm volatile("global_load_async_to_lds_b128 %0, %1, off"
               :: "v"(ldsOff), "v"(g) : "memory");
}
__device__ __forceinline__ void wait_async0() {
  asm volatile("s_wait_asynccnt 0x0" ::: "memory");
}

// ---------------------------------------------------------------------------
// Kernel 1: codebook prep — bf16 hi/lo split + normE[k] = sum(emb[k,:]^2)
// ---------------------------------------------------------------------------
__global__ void vq_prep(const float* __restrict__ emb,
                        __bf16* __restrict__ emb_h,
                        __bf16* __restrict__ emb_l,
                        float* __restrict__ normE) {
  __shared__ float sm[256];
  const int k = blockIdx.x;
  const int d = threadIdx.x;
  const float x = emb[k * D_ + d];
  const __bf16 h = (__bf16)x;
  const __bf16 l = (__bf16)(x - (float)h);
  emb_h[k * D_ + d] = h;
  emb_l[k * D_ + d] = l;
  sm[d] = x * x;
  __syncthreads();
  #pragma unroll
  for (int s = 128; s > 0; s >>= 1) {
    if (d < s) sm[d] += sm[d + s];
    __syncthreads();
  }
  if (d == 0) normE[k] = sm[0];
}

// ---------------------------------------------------------------------------
// Kernel 2: fused distance-GEMM + argmin + gather + loss partials.
// 256 blocks x 8 waves; each wave: 16 rows vs all 8192 codes.
// A register-resident (bf16 hi/lo, WMMA lane layout). B groups (64 codes,
// 64KB hi+lo) staged block-cooperatively into double-buffered LDS via
// global_load_async_to_lds_b128, overlapped with the 96 WMMAs per group.
// ---------------------------------------------------------------------------
__global__ void __launch_bounds__(256, 1)
vq_main(const float* __restrict__ z,
        const float* __restrict__ emb,
        const __bf16* __restrict__ emb_h,
        const __bf16* __restrict__ emb_l,
        const float* __restrict__ normE,
        float* __restrict__ out_q,     // [B,D,T]
        float* __restrict__ out_idx,   // [N] as float
        float* __restrict__ partials)  // [NBLOCKS]
{
  extern __shared__ char dynlds[];
  char* buf0  = dynlds;
  char* buf1  = dynlds + BUF_BYTES;
  int*   s_idx = (int*)(dynlds + 2 * BUF_BYTES);
  float* s_red = (float*)(dynlds + 2 * BUF_BYTES + ROWS_PER_BLOCK * 4);

  const int tid  = threadIdx.x;
  const int lane = tid & 31;
  const int wave = tid >> 5;

  const int n_block = blockIdx.x * ROWS_PER_BLOCK;  // 128 rows, same b
  const int b       = n_block >> 11;                // / T_
  const int t_blk   = n_block & (T_ - 1);
  const int n_wave  = n_block + wave * 16;          // this wave's 16 rows
  const int t_wave  = t_blk + wave * 16;

  // ---- Load A (16 rows x 256 d) of z, convert to bf16 hi/lo, keep in VGPRs.
  // A k order used here (contiguous, same as B): lanes0-15 row M=lane hold K 0..15;
  // lanes16-31 row M=lane-16 hold K 16..31.
  const int mrow  = lane & 15;
  const int half8 = (lane & 16) ? 8 : 0;
  const float* zb = z + (size_t)b * D_ * T_ + (t_wave + mrow);

  v16bf ah[8], al[8];
  #pragma unroll
  for (int c = 0; c < 8; ++c) {
    const int dbase = c * 32;
    v16bf h, l;
    #pragma unroll
    for (int i = 0; i < 8; ++i) {
      const float x0 = zb[(size_t)(dbase + 2 * half8 + i) * T_];
      const float x1 = zb[(size_t)(dbase + 2 * half8 + 8 + i) * T_];
      const __bf16 h0 = (__bf16)x0;
      const __bf16 h1 = (__bf16)x1;
      h[i]     = h0;                     h[8 + i] = h1;
      l[i]     = (__bf16)(x0 - (float)h0);
      l[8 + i] = (__bf16)(x1 - (float)h1);
    }
    ah[c] = h;  al[c] = l;
  }

  // ---- Prologue: async-stage group 0's B (hi+lo) into buf0.
  {
    const char* srcH = (const char*)emb_h;
    const char* srcL = (const char*)emb_l;
    #pragma unroll
    for (int i = 0; i < 8; ++i) {
      async_copy_b128(buf0 + tid * 16 + i * 4096,            srcH + tid * 16 + i * 4096);
      async_copy_b128(buf0 + BUF_HALF + tid * 16 + i * 4096, srcL + tid * 16 + i * 4096);
    }
  }
  wait_async0();
  __syncthreads();

  // ---- Sweep all K codes, 64 (4 tiles of 16) at a time, double-buffered.
  // B k order used here (contiguous, same as A): lanes0-15 col N=lane hold K 0..15 (32B contig);
  // lanes16-31 col N=lane-16 hold K 16..31.
  const int col_in = lane & 15;
  const int koff2  = ((lane & 16) ? 16 : 0) * 2;   // byte offset within row

  float minval[8];
  int   minidx[8];
  #pragma unroll
  for (int v = 0; v < 8; ++v) { minval[v] = 3.4e38f; minidx[v] = 0; }

  #pragma unroll 1
  for (int g = 0; g < NGROUPS; ++g) {
    // Kick off next group's async copy into the other buffer.
    if (g + 1 < NGROUPS) {
      char* nbuf = ((g + 1) & 1) ? buf1 : buf0;
      const char* srcH = (const char*)emb_h + (size_t)(g + 1) * BUF_HALF;
      const char* srcL = (const char*)emb_l + (size_t)(g + 1) * BUF_HALF;
      #pragma unroll
      for (int i = 0; i < 8; ++i) {
        async_copy_b128(nbuf + tid * 16 + i * 4096,            srcH + tid * 16 + i * 4096);
        async_copy_b128(nbuf + BUF_HALF + tid * 16 + i * 4096, srcL + tid * 16 + i * 4096);
      }
    }

    // Compute on current buffer.
    const char* cbuf = (g & 1) ? buf1 : buf0;
    const char* bh_base = cbuf + col_in * 512 + koff2;            // hi rows
    const char* bl_base = cbuf + BUF_HALF + col_in * 512 + koff2; // lo rows

    v8f acc0 = {}, acc1 = {}, acc2 = {}, acc3 = {};
    #pragma unroll
    for (int c = 0; c < 8; ++c) {
      const int co = 64 * c;            // byte offset of 32-d chunk in a row
      const v16bf ahc = ah[c];
      const v16bf alc = al[c];
      const v16bf bh0 = *(const v16bf*)(bh_base + 0 * 16 * 512 + co);
      const v16bf bl0 = *(const v16bf*)(bl_base + 0 * 16 * 512 + co);
      const v16bf bh1 = *(const v16bf*)(bh_base + 1 * 16 * 512 + co);
      const v16bf bl1 = *(const v16bf*)(bl_base + 1 * 16 * 512 + co);
      const v16bf bh2 = *(const v16bf*)(bh_base + 2 * 16 * 512 + co);
      const v16bf bl2 = *(const v16bf*)(bl_base + 2 * 16 * 512 + co);
      const v16bf bh3 = *(const v16bf*)(bh_base + 3 * 16 * 512 + co);
      const v16bf bl3 = *(const v16bf*)(bl_base + 3 * 16 * 512 + co);
      acc0 = __builtin_amdgcn_wmma_f32_16x16x32_bf16(false, ahc, false, bh0, (short)0, acc0, false, false);
      acc1 = __builtin_amdgcn_wmma_f32_16x16x32_bf16(false, ahc, false, bh1, (short)0, acc1, false, false);
      acc2 = __builtin_amdgcn_wmma_f32_16x16x32_bf16(false, ahc, false, bh2, (short)0, acc2, false, false);
      acc3 = __builtin_amdgcn_wmma_f32_16x16x32_bf16(false, ahc, false, bh3, (short)0, acc3, false, false);
      acc0 = __builtin_amdgcn_wmma_f32_16x16x32_bf16(false, alc, false, bh0, (short)0, acc0, false, false);
      acc1 = __builtin_amdgcn_wmma_f32_16x16x32_bf16(false, alc, false, bh1, (short)0, acc1, false, false);
      acc2 = __builtin_amdgcn_wmma_f32_16x16x32_bf16(false, alc, false, bh2, (short)0, acc2, false, false);
      acc3 = __builtin_amdgcn_wmma_f32_16x16x32_bf16(false, alc, false, bh3, (short)0, acc3, false, false);
      acc0 = __builtin_amdgcn_wmma_f32_16x16x32_bf16(false, ahc, false, bl0, (short)0, acc0, false, false);
      acc1 = __builtin_amdgcn_wmma_f32_16x16x32_bf16(false, ahc, false, bl1, (short)0, acc1, false, false);
      acc2 = __builtin_amdgcn_wmma_f32_16x16x32_bf16(false, ahc, false, bl2, (short)0, acc2, false, false);
      acc3 = __builtin_amdgcn_wmma_f32_16x16x32_bf16(false, ahc, false, bl3, (short)0, acc3, false, false);
    }
    const int col0 = g * GROUP_CODES + col_in;
    #pragma unroll
    for (int j = 0; j < 4; ++j) {
      const v8f accj = (j == 0) ? acc0 : (j == 1) ? acc1 : (j == 2) ? acc2 : acc3;
      const int col  = col0 + j * 16;
      const float ne = normE[col];
      #pragma unroll
      for (int v = 0; v < 8; ++v) {
        const float dv = __builtin_fmaf(-2.0f, accj[v], ne);
        if (dv < minval[v]) { minval[v] = dv; minidx[v] = col; }
      }
    }

    // Handoff: next group's copy done + visible to all waves.
    wait_async0();
    __syncthreads();
  }

  // ---- Cross-lane argmin over the 16 columns each half-wave covers.
  // C layout: lane holds rows M = v + 8*(lane/16), col N = lane%16.
  #pragma unroll
  for (int v = 0; v < 8; ++v) {
    #pragma unroll
    for (int m = 1; m < 16; m <<= 1) {
      const float ov = __shfl_xor(minval[v], m, 16);
      const int   oi = __shfl_xor(minidx[v], m, 16);
      if (ov < minval[v] || (ov == minval[v] && oi < minidx[v])) {
        minval[v] = ov; minidx[v] = oi;
      }
    }
  }
  if ((lane & 15) == 0) {          // lanes 0 and 16
    const int rbase = (lane >> 4) * 8;
    #pragma unroll
    for (int v = 0; v < 8; ++v) {
      const int row = rbase + v;
      s_idx[wave * 16 + row] = minidx[v];
      out_idx[n_wave + row]  = (float)minidx[v];
    }
  }
  __syncthreads();

  // ---- Gather quantized rows, write output, accumulate loss partial.
  // quantized_st == quantized numerically; q_loss == e_loss numerically.
  const size_t zbase = (size_t)b * D_ * T_ + t_blk;
  float lsum = 0.0f;
  #pragma unroll 1
  for (int it = 0; it < 128; ++it) {
    const int d  = (it << 1) + (tid >> 7);
    const int tl = tid & 127;
    const float q  = emb[(size_t)s_idx[tl] * D_ + d];
    const size_t a = zbase + (size_t)d * T_ + tl;
    const float zv = z[a];
    out_q[a] = q;
    const float df = q - zv;
    lsum = __builtin_fmaf(df, df, lsum);
  }
  s_red[tid] = lsum;
  __syncthreads();
  #pragma unroll
  for (int s = 128; s > 0; s >>= 1) {
    if (tid < s) s_red[tid] += s_red[tid + s];
    __syncthreads();
  }
  if (tid == 0) partials[blockIdx.x] = s_red[0];
}

// ---------------------------------------------------------------------------
// Kernel 3: deterministic fixed-order loss finalize.
// loss = (1 + 0.25) * sum((q - z)^2) / (B*D*T)
// ---------------------------------------------------------------------------
__global__ void vq_finalize(const float* __restrict__ partials,
                            float* __restrict__ loss_out) {
  if (threadIdx.x == 0) {
    float s = 0.0f;
    for (int i = 0; i < NBLOCKS; ++i) s += partials[i];
    *loss_out = 1.25f * s / (float)QOUT;
  }
}

// ---------------------------------------------------------------------------
extern "C" void kernel_launch(void* const* d_in, const int* in_sizes, int n_in,
                              void* d_out, int out_size, void* d_ws, size_t ws_size,
                              hipStream_t stream) {
  const float* z   = (const float*)d_in[0];  // [B, D, T]
  const float* emb = (const float*)d_in[1];  // [K, D]
  float* out = (float*)d_out;                // quantized | loss | idx

  // Workspace layout (~8.4 MB)
  __bf16* emb_h   = (__bf16*)d_ws;                       // K*D bf16
  __bf16* emb_l   = emb_h + (size_t)K_ * D_;             // K*D bf16
  float*  normE   = (float*)(emb_l + (size_t)K_ * D_);   // K f32
  float*  partial = normE + K_;                          // NBLOCKS f32

  vq_prep<<<K_, 256, 0, stream>>>(emb, emb_h, emb_l, normE);
  vq_main<<<NBLOCKS, 256, LDS_TOTAL, stream>>>(z, emb, emb_h, emb_l, normE,
                                               out, out + QOUT + 1, partial);
  vq_finalize<<<1, 32, 0, stream>>>(partial, out + QOUT);
}
